// SeqAutoEncoder_44487271252522
// MI455X (gfx1250) — matrix operand order repaired and checked by reading, not compiled, not run on hardware
//
#include <hip/hip_runtime.h>

// ---------------- CDNA5 (gfx1250) types ----------------
typedef __attribute__((ext_vector_type(16))) __bf16 v16bf;
typedef __attribute__((ext_vector_type(8)))  float  v8f;

#define Bdim    128
#define Tdim    2048
#define Hdim    64
#define G4      256      // 4*H
#define ROWS    16       // batch rows per workgroup (== WMMA M)
#define NTHREADS 512

union BfVec { v16bf v; unsigned short s[16]; };

// float -> bf16 bits, round-to-nearest-even
__device__ inline unsigned short f2bf(float x) {
  union { float f; unsigned int u; } v; v.f = x;
  unsigned int r = v.u + 0x7FFFu + ((v.u >> 16) & 1u);
  return (unsigned short)(r >> 16);
}

// Branch-free activations: one v_exp_f32 + one v_rcp_f32 each.
__device__ inline float sigm_(float x) {
  return __builtin_amdgcn_rcpf(1.0f + __expf(-x));
}
__device__ inline float tanh_(float x) {
  // tanh(x) = 1 - 2/(1 + e^{2x}); saturates correctly at +/-1 on overflow.
  return 1.0f - 2.0f * __builtin_amdgcn_rcpf(1.0f + __expf(2.0f * x));
}

// A-fragment (16-bit, 16x32 per K-chunk) swizzled LDS offset for element (m, k),
// in contiguous K order (k%32 = 16*(lane/16) + e), matching the B fragments. Stored so each lane's 16
// fragment elements are contiguous (2x ds_load_b128 per chunk).
__device__ inline int afrag_off(int m, int k) {
  int kc      = k >> 5;
  int kl      = k & 31;
  int within  = kl & 15;
  int hi16    = kl >> 4;                 // k%32>=16 -> lanes 16..31
  int lanegrp = hi16;
  int e       = within;
  int lane    = m + (lanegrp << 4);
  return ((kc << 5) + lane) * 16 + e;    // units: bf16 elements
}

// B fragment: col N = lane%16, K = (lane<16?0:16)+e within each 32-K chunk.
__device__ inline void load_gate_wfrags(const float* __restrict__ Wih,
                                        const float* __restrict__ Whh,
                                        int wave, int lane, v16bf wfrag[4]) {
  int col   = wave * 16 + (lane & 15);   // gate column 0..255
  int kbase = (lane & 16) ? 16 : 0;
#pragma unroll
  for (int kc = 0; kc < 4; ++kc) {
    BfVec wb;
#pragma unroll
    for (int e = 0; e < 16; ++e) {
      int k   = (kc << 5) + kbase + e;   // 0..127: x features then h features
      float w = (k < 64) ? Wih[col * 64 + k] : Whh[col * 64 + (k - 64)];
      wb.s[e] = f2bf(w);
    }
    wfrag[kc] = wb.v;
  }
}

// Load x(t) for this WG's 16 rows into A-fragment chunks 0..1 (2 elems/thread).
__device__ inline void load_x_step(const float* __restrict__ x, int b0, int t,
                                   unsigned short* act16, int tid) {
  int item = tid * 2;
  {
    int b = item >> 6, j = item & 63;
    size_t gidx = ((size_t)(b0 + b) * Tdim + t) * 64 + j;
    __builtin_prefetch(&x[gidx + 64], 0, 3);   // t+1 row (speculative, WGP-near)
    act16[afrag_off(b, j)]     = f2bf(x[gidx]);
    act16[afrag_off(b, j + 1)] = f2bf(x[gidx + 1]);  // j even: same cacheline
  }
}

// FC on h resident in act16 chunks 2..3 -> x_recon[t_out] (waves 0..3 only).
// fc bias is pre-folded into the C operand of the first WMMA.
__device__ inline void fc_step(const unsigned short* act16, const v16bf fcfrag[2],
                               v8f fcbacc, float* __restrict__ out,
                               int b0, int t_out, int wave, int lane) {
  v16bf a0 = *(const v16bf*)&act16[((2 << 5) + lane) * 16];
  v16bf a1 = *(const v16bf*)&act16[((3 << 5) + lane) * 16];
  v8f f = __builtin_amdgcn_wmma_f32_16x16x32_bf16(
      false, a0, false, fcfrag[0], (short)0, fcbacc, false, false);
  f = __builtin_amdgcn_wmma_f32_16x16x32_bf16(
      false, a1, false, fcfrag[1], (short)0, f, false, false);
  int m    = lane & 15;
  int dgrp = (lane & 16) ? 8 : 0;
  int n    = wave * 16 + m;
#pragma unroll
  for (int r = 0; r < 8; ++r)
    out[((size_t)(b0 + r + dgrp) * Tdim + t_out) * 64 + n] = f[r];
}

// One LSTM scan phase (2048 steps) for 16 batch rows; 2 barriers per step.
// Cell state c lives in registers (creg), bias in the WMMA C operand.
template<bool IS_DEC>
__device__ void lstm_phase(const float* __restrict__ x,        // [B,T,64]
                           const float* __restrict__ Wih,
                           const float* __restrict__ Whh,
                           const float* __restrict__ bih,
                           const float* __restrict__ bhh,
                           unsigned short* act16, float* gatesBuf,
                           float creg[2],
                           const v16bf fcfrag[2], v8f fcbacc,
                           float* __restrict__ out,            // x_recon [B,T,64]
                           float* __restrict__ henc_out,       // [B,64]
                           int b0, int tid, int wave, int lane) {
  v16bf wfrag[4];
  load_gate_wfrags(Wih, Whh, wave, lane, wfrag);
  const int m    = lane & 15;
  const int dgrp = (lane & 16) ? 8 : 0;  // D rows: M = r + dgrp
  const int gcol = wave * 16 + m;

  // bias for this wave's gate column, broadcast into a loop-invariant v8f
  float bb = bih[gcol] + bhh[gcol];
  v8f biasacc = {bb, bb, bb, bb, bb, bb, bb, bb};

  // preload x(0); act16 chunks 2..3 already hold h_init
  load_x_step(x, b0, 0, act16, tid);
  __syncthreads();

  for (int t = 0; t < Tdim; ++t) {
    // ---- stage A: gates(t) = bias + [x_t || h_{t-1}] x Wcat (bf16 WMMA) ----
    v16bf a0 = *(const v16bf*)&act16[((0 << 5) + lane) * 16];
    v16bf a1 = *(const v16bf*)&act16[((1 << 5) + lane) * 16];
    v16bf a2 = *(const v16bf*)&act16[((2 << 5) + lane) * 16];
    v16bf a3 = *(const v16bf*)&act16[((3 << 5) + lane) * 16];
    v8f acc = __builtin_amdgcn_wmma_f32_16x16x32_bf16(
        false, a0, false, wfrag[0], (short)0, biasacc, false, false);
    acc = __builtin_amdgcn_wmma_f32_16x16x32_bf16(
        false, a1, false, wfrag[1], (short)0, acc, false, false);
    acc = __builtin_amdgcn_wmma_f32_16x16x32_bf16(
        false, a2, false, wfrag[2], (short)0, acc, false, false);
    acc = __builtin_amdgcn_wmma_f32_16x16x32_bf16(
        false, a3, false, wfrag[3], (short)0, acc, false, false);
    // pipelined FC: output step t-1 (h_{t-1} resident; read-read with gates)
    if (IS_DEC && t > 0) {
      if (wave < 4) fc_step(act16, fcfrag, fcbacc, out, b0, t - 1, wave, lane);
    }
#pragma unroll
    for (int r = 0; r < 8; ++r)
      gatesBuf[(r + dgrp) * G4 + gcol] = acc[r];
    __syncthreads();

    // ---- stage B: cell update (f32, branch-free, c in registers) ----
    {
      int item = tid * 2;
#pragma unroll
      for (int q = 0; q < 2; ++q, ++item) {
        int b = item >> 6, j = item & 63;
        const float* gb = &gatesBuf[b * G4];
        float gi = gb[j];
        float gf = gb[64 + j];
        float gg = gb[128 + j];
        float go = gb[192 + j];
        float c  = sigm_(gf) * creg[q] + sigm_(gi) * tanh_(gg);
        float h  = sigm_(go) * tanh_(c);
        creg[q] = c;
        act16[afrag_off(b, 64 + j)] = f2bf(h);   // feeds next step's A chunks 2,3
        if (!IS_DEC && t == Tdim - 1)
          henc_out[(size_t)(b0 + b) * 64 + j] = h;
      }
    }
    if (t + 1 < Tdim) load_x_step(x, b0, t + 1, act16, tid);
    __syncthreads();
  }

  // tail: FC for the final decoder step (h_{T-1} resident after last barrier)
  if (IS_DEC) {
    if (wave < 4) fc_step(act16, fcfrag, fcbacc, out, b0, Tdim - 1, wave, lane);
  }
}

__global__ __launch_bounds__(NTHREADS, 1)
void seqae_persistent_kernel(const float* __restrict__ x,
                             const float* __restrict__ x_enc,
                             const float* __restrict__ eWih, const float* __restrict__ eWhh,
                             const float* __restrict__ ebih, const float* __restrict__ ebhh,
                             const float* __restrict__ dWih, const float* __restrict__ dWhh,
                             const float* __restrict__ dbih, const float* __restrict__ dbhh,
                             const float* __restrict__ fcW,  const float* __restrict__ fcb,
                             float* __restrict__ d_out) {
  __shared__ __align__(32) unsigned short act16[4 * 32 * 16]; // 4KB, A-frag layout
  __shared__ float gatesBuf[ROWS * G4];                       // 16KB

  const int tid  = threadIdx.x;
  const int wave = tid >> 5;
  const int lane = tid & 31;
  const int b0   = blockIdx.x * ROWS;

  // FC weight fragments + bias accumulator (waves 0..3)
  v16bf fcfrag[2];
  v8f fcbacc = {0.f, 0.f, 0.f, 0.f, 0.f, 0.f, 0.f, 0.f};
  {
    BfVec z;
#pragma unroll
    for (int e = 0; e < 16; ++e) z.s[e] = 0;
    fcfrag[0] = z.v; fcfrag[1] = z.v;
    if (wave < 4) {
      int col   = wave * 16 + (lane & 15);    // output feature n, 0..63
      int kbase = (lane & 16) ? 16 : 0;
#pragma unroll
      for (int kc = 0; kc < 2; ++kc) {
        BfVec wb;
#pragma unroll
        for (int e = 0; e < 16; ++e) {
          int k = (kc << 5) + kbase + e;      // input feature j
          wb.s[e] = f2bf(fcW[col * 64 + k]);  // fc_W[n][j]
        }
        fcfrag[kc] = wb.v;
      }
      float fb = fcb[col];
      fcbacc = (v8f){fb, fb, fb, fb, fb, fb, fb, fb};
    }
  }

  // zero initial state: c (registers) = 0, h (bf16 in act16 chunks 2,3) = 0
  float creg[2] = {0.0f, 0.0f};
  {
    int item = tid * 2;
#pragma unroll
    for (int q = 0; q < 2; ++q, ++item) {
      int b = item >> 6, j = item & 63;
      act16[afrag_off(b, 64 + j)] = 0;
    }
  }
  __syncthreads();

  float* henc_out = d_out + (size_t)Bdim * Tdim * 64;

  // encoder: discards per-step h, leaves (h,c) in act16/creg, writes h_enc
  lstm_phase<false>(x, eWih, eWhh, ebih, ebhh, act16, gatesBuf, creg,
                    fcfrag, fcbacc, nullptr, henc_out, b0, tid, wave, lane);

  // decoder: initial (h,c) = encoder final state (already resident), fused FC
  lstm_phase<true>(x_enc, dWih, dWhh, dbih, dbhh, act16, gatesBuf, creg,
                   fcfrag, fcbacc, d_out, nullptr, b0, tid, wave, lane);
}

extern "C" void kernel_launch(void* const* d_in, const int* in_sizes, int n_in,
                              void* d_out, int out_size, void* d_ws, size_t ws_size,
                              hipStream_t stream) {
  (void)in_sizes; (void)n_in; (void)out_size; (void)d_ws; (void)ws_size;
  const float* x     = (const float*)d_in[0];
  const float* x_enc = (const float*)d_in[1];
  const float* eWih  = (const float*)d_in[2];
  const float* eWhh  = (const float*)d_in[3];
  const float* ebih  = (const float*)d_in[4];
  const float* ebhh  = (const float*)d_in[5];
  const float* dWih  = (const float*)d_in[6];
  const float* dWhh  = (const float*)d_in[7];
  const float* dbih  = (const float*)d_in[8];
  const float* dbhh  = (const float*)d_in[9];
  const float* fcW   = (const float*)d_in[10];
  const float* fcb   = (const float*)d_in[11];
  float* out = (float*)d_out;

  dim3 grid(Bdim / ROWS);   // 8 workgroups, one per 16 batch rows
  dim3 block(NTHREADS);     // 16 waves of 32
  seqae_persistent_kernel<<<grid, block, 0, stream>>>(
      x, x_enc, eWih, eWhh, ebih, ebhh, dWih, dWhh, dbih, dbhh, fcW, fcb, out);
}
